// ProbSparseAttention_32195074851273
// MI455X (gfx1250) — matrix operand order repaired and checked by reading, not compiled, not run on hardware
//
#include <hip/hip_runtime.h>

#define BATCH 8
#define SEQ   2048
#define DIM   512
#define KTOP  204                          // int(2048 * 0.1)
#define SCALE 0.044194173824159216f        // 1/sqrt(512)

typedef __attribute__((ext_vector_type(16))) __bf16 v16bf;
typedef __attribute__((ext_vector_type(8)))  float  v8f;

struct alignas(16) U4 { unsigned x, y, z, w; };
union AB { v16bf v; U4 q[2]; };

// monotonic float -> uint transform (order-preserving, for descending radix select)
__device__ __forceinline__ unsigned ford(float f) {
    unsigned u = __float_as_uint(f);
    return (u & 0x80000000u) ? ~u : (u | 0x80000000u);
}

// ---------------------------------------------------------------------------
// Pre-pass: fp32 -> bf16 conversion (Q, K)
// ---------------------------------------------------------------------------
__global__ void cvt_bf16_kernel(const float* __restrict__ src,
                                __bf16* __restrict__ dst, int n) {
    for (int i = blockIdx.x * blockDim.x + threadIdx.x; i < n;
         i += gridDim.x * blockDim.x)
        dst[i] = (__bf16)src[i];
}

// Pre-pass: V [B][S][D] fp32 -> Vt [B][D][S] bf16 (so WMMA B-operand columns
// of V are contiguous along the seq/contraction axis)
__global__ void transpose_v_kernel(const float* __restrict__ V,
                                   __bf16* __restrict__ Vt) {
    const int total = BATCH * SEQ * DIM;
    for (int i = blockIdx.x * blockDim.x + threadIdx.x; i < total;
         i += gridDim.x * blockDim.x) {
        int d = i % DIM;
        int s = (i / DIM) % SEQ;
        int b = i / (DIM * SEQ);
        Vt[((size_t)(b * DIM + d)) * SEQ + s] = (__bf16)V[i];
    }
}

// ---------------------------------------------------------------------------
// Main kernel: one workgroup = one batch b, one 16-row query tile.
// 256 threads = 8 wave32.
// ---------------------------------------------------------------------------
__global__ __launch_bounds__(256) void probsparse_attn(
    const __bf16* __restrict__ Qb, const __bf16* __restrict__ Kb,
    const __bf16* __restrict__ Vt, float* __restrict__ out) {

    extern __shared__ char smem[];
    float*    sc   = (float*)smem;                            // [16][SEQ] scores
    __bf16*   pB   = (__bf16*)(smem + 16 * SEQ * 4);          // [16][SEQ] probs
    unsigned* hist = (unsigned*)(smem + 16 * SEQ * 6);        // [8][256]
    unsigned* uthr = (unsigned*)(smem + 16 * SEQ * 6 + 8192); // [16]
    float*    rmax = (float*)(smem + 16 * SEQ * 6 + 8192 + 64);
    float*    den  = (float*)(smem + 16 * SEQ * 6 + 8192 + 128);

    const int tid  = threadIdx.x;
    const int lane = tid & 31;
    const int wv   = tid >> 5;       // wave id 0..7
    const int half = lane >> 4;      // 0 or 1
    const int l16  = lane & 15;
    const int b    = blockIdx.y;
    const int q0   = blockIdx.x * 16;

    if (tid < 16) den[tid] = 0.0f;

    // ---- Phase 1: scores(16 x 2048) = Q(16x512) * K^T, WMMA bf16 ----------
    // A (Q) tiles preloaded once into registers (16 x v16bf = 128 VGPRs).
    // B (K) tiles stream through a depth-4 register ring so global_load_b128
    // for dk+3 issues before the WMMA for dk (overlaps L2 latency).
    const __bf16* Qrow = Qb + ((size_t)(b * SEQ + q0 + l16)) * DIM;
    AB aT[16];
#pragma unroll
    for (int dk = 0; dk < 16; ++dk) {
        aT[dk].q[0] = *(const U4*)(Qrow + dk * 32 + half * 16);
        aT[dk].q[1] = *(const U4*)(Qrow + dk * 32 + half * 16 + 8);
    }

    for (int t = 0; t < 16; ++t) {
        const int col0 = wv * 256 + t * 16;
        const __bf16* Krow = Kb + ((size_t)(b * SEQ + col0 + l16)) * DIM;
        if (t < 15) __builtin_prefetch(Krow + 16 * DIM, 0, 1);

        AB bm[4];
#pragma unroll
        for (int i = 0; i < 3; ++i) {          // prime 3 tiles ahead
            bm[i].q[0] = *(const U4*)(Krow + i * 32 + half * 16);
            bm[i].q[1] = *(const U4*)(Krow + i * 32 + half * 16 + 8);
        }
        v8f acc = {0.f, 0.f, 0.f, 0.f, 0.f, 0.f, 0.f, 0.f};
#pragma unroll
        for (int dk = 0; dk < 16; ++dk) {
            if (dk < 13) {
                const int kb = (dk + 3) * 32;
                bm[(dk + 3) & 3].q[0] = *(const U4*)(Krow + kb + half * 16);
                bm[(dk + 3) & 3].q[1] = *(const U4*)(Krow + kb + half * 16 + 8);
            }
            acc = __builtin_amdgcn_wmma_f32_16x16x32_bf16(
                false, aT[dk].v, false, bm[dk & 3].v, (short)0, acc, false,
                false);
        }
#pragma unroll
        for (int r = 0; r < 8; ++r)
            sc[(r + half * 8) * SEQ + col0 + l16] = acc[r] * SCALE;
    }
    __syncthreads();

    // ---- Phase 2: per-row max + exact kth-largest via 4-pass radix select --
    unsigned* myh = hist + wv * 256;
    for (int rr = 0; rr < 2; ++rr) {
        const int row = wv * 2 + rr;
        const float* srow = sc + row * SEQ;

        float m = -3.0e38f;
        for (int j = lane; j < SEQ; j += 32) m = fmaxf(m, srow[j]);
        for (int off = 16; off > 0; off >>= 1)
            m = fmaxf(m, __shfl_xor(m, off, 32));

        unsigned prefix = 0, mask = 0;
        int need = KTOP;
        for (int p = 3; p >= 0; --p) {
            const int sh = p * 8;
            for (int i = lane; i < 256; i += 32) myh[i] = 0;
            for (int j = lane; j < SEQ; j += 32) {
                unsigned u = ford(srow[j]);
                if ((u & mask) == prefix)
                    atomicAdd(&myh[(u >> sh) & 255u], 1u);
            }
            // all lanes redundantly walk bins (uniform, LDS broadcast reads)
            unsigned accum = 0;
            int d = 255;
            for (; d > 0; --d) {
                unsigned c = myh[d];
                if (accum + c >= (unsigned)need) break;
                accum += c;
            }
            need -= (int)accum;
            prefix |= ((unsigned)d) << sh;
            mask   |= 0xFFu << sh;
        }
        if (lane == 0) { uthr[row] = prefix; rmax[row] = fmaxf(m, 0.0f); }
    }
    __syncthreads();

    // ---- Phase 3: P = exp(sparse - m) as bf16; row denominators -----------
    {
        const int row = tid >> 4;
        const int c   = tid & 15;
        const unsigned th = uthr[row];
        const float m = rmax[row];
        const float* srow = sc + row * SEQ;
        __bf16* prow = pB + row * SEQ;
        float psum = 0.0f;
        for (int j = c; j < SEQ; j += 16) {
            float s = srow[j];
            float v = (ford(s) >= th) ? s : 0.0f;  // non-topk entries are 0
            float p = __expf(v - m);
            psum += p;
            prow[j] = (__bf16)p;
        }
        atomicAdd(&den[row], psum);
    }
    __syncthreads();

    // ---- Phase 4: out(16 x 512) = P(16x2048) * V(2048x512), WMMA bf16 -----
    // Double-buffered across jt: next iteration's LDS A-tile and 4 global
    // Vt B-tiles load while the current 4 WMMAs execute.
    const int dcol0 = wv * 64;       // each wave owns 4 n-tiles of 16 cols
    const __bf16* prow0 = pB + l16 * SEQ;
    const __bf16* vbase =
        Vt + ((size_t)(b * DIM + dcol0 + l16)) * SEQ + half * 16;

    v8f oacc[4] = {};
    AB a2[2], bmb[2][4];

    a2[0].q[0] = *(const U4*)(prow0 + half * 16);
    a2[0].q[1] = *(const U4*)(prow0 + half * 16 + 8);
#pragma unroll
    for (int nt = 0; nt < 4; ++nt) {
        const __bf16* vp = vbase + (size_t)(nt * 16) * SEQ;
        bmb[0][nt].q[0] = *(const U4*)(vp);
        bmb[0][nt].q[1] = *(const U4*)(vp + 8);
    }

#pragma unroll 2
    for (int jt = 0; jt < 64; ++jt) {
        const int cur = jt & 1, nxt = cur ^ 1;
        if (jt < 63) {
            const int j1 = (jt + 1) * 32;
            a2[nxt].q[0] = *(const U4*)(prow0 + j1 + half * 16);
            a2[nxt].q[1] = *(const U4*)(prow0 + j1 + half * 16 + 8);
#pragma unroll
            for (int nt = 0; nt < 4; ++nt) {
                const __bf16* vp = vbase + (size_t)(nt * 16) * SEQ + j1;
                bmb[nxt][nt].q[0] = *(const U4*)(vp);
                bmb[nxt][nt].q[1] = *(const U4*)(vp + 8);
            }
        }
#pragma unroll
        for (int nt = 0; nt < 4; ++nt)
            oacc[nt] = __builtin_amdgcn_wmma_f32_16x16x32_bf16(
                false, a2[cur].v, false, bmb[cur][nt].v, (short)0, oacc[nt],
                false, false);
    }

#pragma unroll
    for (int nt = 0; nt < 4; ++nt) {
#pragma unroll
        for (int r = 0; r < 8; ++r) {
            const int M = r + half * 8;
            const float inv = 1.0f / den[M];
            out[((size_t)(b * SEQ + q0 + M)) * DIM + dcol0 + nt * 16 + l16] =
                oacc[nt][r] * inv;
        }
    }
}

// ---------------------------------------------------------------------------
extern "C" void kernel_launch(void* const* d_in, const int* in_sizes, int n_in,
                              void* d_out, int out_size, void* d_ws,
                              size_t ws_size, hipStream_t stream) {
    const float* Q = (const float*)d_in[0];
    const float* K = (const float*)d_in[1];
    const float* V = (const float*)d_in[2];

    const size_t n = (size_t)BATCH * SEQ * DIM;   // 8.39M elements
    __bf16* Qb = (__bf16*)d_ws;                   // n bf16
    __bf16* Kb = Qb + n;                          // n bf16
    __bf16* Vt = Kb + n;                          // n bf16 (transposed V)

    cvt_bf16_kernel<<<2048, 256, 0, stream>>>(Q, Qb, (int)n);
    cvt_bf16_kernel<<<2048, 256, 0, stream>>>(K, Kb, (int)n);
    transpose_v_kernel<<<2048, 256, 0, stream>>>(V, Vt);

    const int smemBytes = 16 * SEQ * 6 + 8192 + 192;   // 204,992 B dynamic LDS
    hipFuncSetAttribute((const void*)probsparse_attn,
                        hipFuncAttributeMaxDynamicSharedMemorySize, smemBytes);

    probsparse_attn<<<dim3(SEQ / 16, BATCH), 256, smemBytes, stream>>>(
        Qb, Kb, Vt, (float*)d_out);
}
